// XNORLinear_91250875170857
// MI455X (gfx1250) — compile-verified, matrix operand order checked by reading, not run on hardware
//
#include <hip/hip_runtime.h>
#include <stdint.h>

// ---------------------------------------------------------------------------
// XNOR linear: out = sign(x) @ sign(W)^T, M=N=K=4096, fp32 out.
// Pass 1: binarize to FP8 E4M3 (+1.0=0x38, -1.0=0xB8, 0=0x00 -- exact).
// Pass 2: FP8 WMMA GEMM (v_wmma_f32_16x16x128_fp8_fp8), double-buffered LDS
//         filled by GLOBAL_LOAD_ASYNC_TO_LDS_B128 (ASYNCcnt-tracked).
// ---------------------------------------------------------------------------

typedef int   v4i  __attribute__((ext_vector_type(4)));
typedef int   v16i __attribute__((ext_vector_type(16)));
typedef float v8f  __attribute__((ext_vector_type(8)));

#define GEMM_M 4096
#define GEMM_N 4096
#define GEMM_K 4096

#define TILE_M 128          // per workgroup
#define TILE_N 128          // per workgroup
#define TILE_K 128          // K per WMMA step
#define NUM_KSTEPS (GEMM_K / TILE_K)
#define LDS_STRIDE 144      // 128 + 16 pad: 16B-aligned, conflict-free b64 reads

#define AS1 __attribute__((address_space(1)))
#define AS3 __attribute__((address_space(3)))

#if defined(__has_builtin)
#if __has_builtin(__builtin_amdgcn_global_load_async_to_lds_b128) && \
    __has_builtin(__builtin_amdgcn_s_wait_asynccnt)
#define USE_ASYNC_LDS 1
#endif
#endif
#ifndef USE_ASYNC_LDS
#define USE_ASYNC_LDS 0
#endif

// ---------------------------------------------------------------------------
// Pass 1: sign-binarize fp32 -> fp8 e4m3 bytes. 16 floats in, 16 bytes out.
// ---------------------------------------------------------------------------
__global__ __launch_bounds__(256) void binarize_fp8_kernel(
    const float* __restrict__ in, uint8_t* __restrict__ out)
{
    const int idx = blockIdx.x * 256 + threadIdx.x;
    const float4* src = reinterpret_cast<const float4*>(in) + (size_t)idx * 4;

    uint32_t words[4];
#pragma unroll
    for (int i = 0; i < 4; ++i) {
        float4 f = src[i];
        float v[4] = {f.x, f.y, f.z, f.w};
        uint32_t w = 0;
#pragma unroll
        for (int j = 0; j < 4; ++j) {
            uint32_t b = (v[j] > 0.0f) ? 0x38u : ((v[j] < 0.0f) ? 0xB8u : 0x00u);
            w |= b << (8 * j);
        }
        words[i] = w;
    }
    reinterpret_cast<uint4*>(out)[idx] =
        make_uint4(words[0], words[1], words[2], words[3]);
}

// ---------------------------------------------------------------------------
// Pass 2: FP8 WMMA GEMM.
// WG: 256 threads = 8 waves (4 in M x 2 in N), tile 128x128, K-steps of 128.
// Each wave: 32(M) x 64(N) patch = 2 A-frags x 4 B-frags = 8 WMMAs / K-step.
// Double-buffered LDS, async global->LDS fill.
// ---------------------------------------------------------------------------
__global__ __launch_bounds__(256) void xnor_gemm_fp8_kernel(
    const uint8_t* __restrict__ A,   // [4096,4096] fp8, K contiguous (sign(x))
    const uint8_t* __restrict__ B,   // [4096,4096] fp8, K contiguous (sign(W))
    float* __restrict__ C)           // [4096,4096] f32
{
    __shared__ __align__(16) uint8_t lsA[2][TILE_M * LDS_STRIDE];
    __shared__ __align__(16) uint8_t lsB[2][TILE_N * LDS_STRIDE];

    const int tid   = threadIdx.x;
    const int lane  = tid & 31;
    const int wave  = tid >> 5;      // 0..7
    const int waveM = wave & 3;      // 0..3 -> 32-row slab
    const int waveN = wave >> 2;     // 0..1 -> 64-col slab
    const int half  = lane >> 4;     // 0/1
    const int r     = lane & 15;     // row/col within 16x16 tile

    const int m0 = blockIdx.y * TILE_M;
    const int n0 = blockIdx.x * TILE_N;

    // cooperative fill: thread -> (row slice, 16B chunk); 4 chunks each for A,B
    const int ldRow   = tid >> 3;        // 0..31
    const int ldChunk = (tid & 7) * 16;  // byte offset within 128B row

    // stage loader: fills lsA[buf], lsB[buf] with K-slice [kk, kk+128)
    auto load_stage = [&](int buf, int kk) {
#pragma unroll
        for (int i = 0; i < 4; ++i) {
            const int row = ldRow + i * 32;
            const uint8_t* gA = A + (size_t)(m0 + row) * GEMM_K + kk + ldChunk;
            const uint8_t* gB = B + (size_t)(n0 + row) * GEMM_K + kk + ldChunk;
            uint8_t* lA = &lsA[buf][row * LDS_STRIDE + ldChunk];
            uint8_t* lB = &lsB[buf][row * LDS_STRIDE + ldChunk];
#if USE_ASYNC_LDS
            __builtin_amdgcn_global_load_async_to_lds_b128(
                (AS1 v4i*)gA, (AS3 v4i*)lA, 0, 0);
            __builtin_amdgcn_global_load_async_to_lds_b128(
                (AS1 v4i*)gB, (AS3 v4i*)lB, 0, 0);
#else
            *reinterpret_cast<uint4*>(lA) =
                *reinterpret_cast<const uint4*>(gA);
            *reinterpret_cast<uint4*>(lB) =
                *reinterpret_cast<const uint4*>(gB);
#endif
        }
    };

    v8f acc[2][4];
#pragma unroll
    for (int tm = 0; tm < 2; ++tm)
#pragma unroll
        for (int tn = 0; tn < 4; ++tn)
#pragma unroll
            for (int j = 0; j < 8; ++j)
                acc[tm][tn][j] = 0.0f;

    load_stage(0, 0);

    for (int k = 0; k < NUM_KSTEPS; ++k) {
        const int cur = k & 1;

        if (k + 1 < NUM_KSTEPS) {
            load_stage(cur ^ 1, (k + 1) * TILE_K);
#if USE_ASYNC_LDS
            // 8 loads just issued for the next stage; wait for current stage
            __builtin_amdgcn_s_wait_asynccnt(8);
#endif
        } else {
#if USE_ASYNC_LDS
            __builtin_amdgcn_s_wait_asynccnt(0);
#endif
        }
        __syncthreads();   // current stage visible to all waves

        // --- fragment loads from LDS ---------------------------------------
        // 8-bit 16x128 operand layout (ISA 7.12.2): lane (r, half) holds
        // bytes at offsets half*8 + 16*j (j=0..7) from its row base.
        v16i afrag[2], bfrag[4];
#pragma unroll
        for (int t = 0; t < 2; ++t) {
            const uint8_t* base =
                &lsA[cur][(waveM * 32 + t * 16 + r) * LDS_STRIDE + half * 8];
            union { v16i v; unsigned long long q[8]; } u;
#pragma unroll
            for (int j = 0; j < 8; ++j)
                u.q[j] = *reinterpret_cast<const unsigned long long*>(base + j * 16);
            afrag[t] = u.v;
        }
#pragma unroll
        for (int t = 0; t < 4; ++t) {
            const uint8_t* base =
                &lsB[cur][(waveN * 64 + t * 16 + r) * LDS_STRIDE + half * 8];
            union { v16i v; unsigned long long q[8]; } u;
#pragma unroll
            for (int j = 0; j < 8; ++j)
                u.q[j] = *reinterpret_cast<const unsigned long long*>(base + j * 16);
            bfrag[t] = u.v;
        }

        // --- 2x4 WMMA tiles, K += 128 --------------------------------------
#pragma unroll
        for (int tm = 0; tm < 2; ++tm)
#pragma unroll
            for (int tn = 0; tn < 4; ++tn)
                acc[tm][tn] = __builtin_amdgcn_wmma_f32_16x16x128_fp8_fp8(
                    afrag[tm], bfrag[tn], (short)0, acc[tm][tn],
                    /*reuse_a=*/false, /*reuse_b=*/false);

        __syncthreads();   // all reads of 'cur' done before it is refilled
    }

    // epilogue: C/D layout -> VGPR j, lane: M = j + half*8, N = r
#pragma unroll
    for (int tm = 0; tm < 2; ++tm) {
#pragma unroll
        for (int tn = 0; tn < 4; ++tn) {
#pragma unroll
            for (int j = 0; j < 8; ++j) {
                const int row = m0 + waveM * 32 + tm * 16 + half * 8 + j;
                const int col = n0 + waveN * 64 + tn * 16 + r;
                __builtin_nontemporal_store(acc[tm][tn][j],
                                            C + (size_t)row * GEMM_N + col);
            }
        }
    }
}

// ---------------------------------------------------------------------------
extern "C" void kernel_launch(void* const* d_in, const int* in_sizes, int n_in,
                              void* d_out, int out_size, void* d_ws, size_t ws_size,
                              hipStream_t stream)
{
    const float* x = reinterpret_cast<const float*>(d_in[0]);  // [4096,4096]
    const float* W = reinterpret_cast<const float*>(d_in[1]);  // [4096,4096]
    float* out = reinterpret_cast<float*>(d_out);

    uint8_t* Abin = reinterpret_cast<uint8_t*>(d_ws);
    uint8_t* Bbin = Abin + (size_t)GEMM_M * GEMM_K;  // 16 MB each

    const int n_vec16 = (GEMM_M * GEMM_K) / 16;
    binarize_fp8_kernel<<<n_vec16 / 256, 256, 0, stream>>>(x, Abin);
    binarize_fp8_kernel<<<n_vec16 / 256, 256, 0, stream>>>(W, Bbin);

    dim3 grid(GEMM_N / TILE_N, GEMM_M / TILE_M);  // (32, 32)
    xnor_gemm_fp8_kernel<<<grid, 256, 0, stream>>>(Abin, Bbin, out);
}
